// StructuralInfoNCELoss_40226663694517
// MI455X (gfx1250) — matrix operand order repaired and checked by reading, compiled, not run on hardware
//
#include <hip/hip_runtime.h>

// ---------------------------------------------------------------------------
// StructuralInfoNCE loss, MI455X (gfx1250, wave32, WMMA).
//
//   G = sum_f X_f X_f^T   (X_f : [512, 16384] fp32)   -- bf16 hi/lo split WMMA
//   sims_ij = -(G_ii + G_jj - 2 G_ij) / (16384 * 3 * 0.1)
//   loss = mean over rows-with-positives of (lse_mask - lse_pos)
// ---------------------------------------------------------------------------

#define B_N    512
#define KDIM   16384
#define TILE   128
#define KT     32
#define KSLICE 1024
#define NSLICE (KDIM / KSLICE)          // 16 K-slices per feature
#define SCALE  (1.0f / (16384.0f * 3.0f * 0.1f))

typedef __attribute__((ext_vector_type(16))) __bf16         v16bf;
typedef __attribute__((ext_vector_type(8)))  float          v8f;
typedef __attribute__((ext_vector_type(8)))  unsigned short u16x8;

union Frag {
  v16bf bf;
  u16x8 us[2];
};

__device__ __forceinline__ unsigned short bf16_rne(float f) {
  unsigned u = __float_as_uint(f);
  unsigned r = 0x7FFFu + ((u >> 16) & 1u);
  return (unsigned short)((u + r) >> 16);
}

__global__ void zero_ws(float* __restrict__ p, int n) {
  int i = blockIdx.x * blockDim.x + threadIdx.x;
  if (i < n) p[i] = 0.0f;
}

// ---------------------------------------------------------------------------
// Gram matrix via bf16 WMMA with hi/lo error compensation.
// grid = (4, 4, 3*NSLICE), block = 256 (8 waves).
// ---------------------------------------------------------------------------
__global__ __launch_bounds__(256) void gram_wmma(
    const float* __restrict__ x0, const float* __restrict__ x1,
    const float* __restrict__ x2, float* __restrict__ G)
{
  __shared__ __align__(16) unsigned short Ahi[TILE * KT];
  __shared__ __align__(16) unsigned short Alo[TILE * KT];
  __shared__ __align__(16) unsigned short Bhi[TILE * KT];
  __shared__ __align__(16) unsigned short Blo[TILE * KT];

  const int z     = blockIdx.z;
  const float* X  = (z < NSLICE) ? x0 : (z < 2 * NSLICE) ? x1 : x2;
  const int kbase = (z & (NSLICE - 1)) * KSLICE;
  const int tileM = blockIdx.y * TILE;
  const int tileN = blockIdx.x * TILE;

  const int t      = threadIdx.x;
  const int lane   = t & 31;
  const int w      = t >> 5;
  const int waveM0 = (w & 1) * 64;     // 2 waves along M -> 64 rows each
  const int waveN0 = (w >> 1) * 32;    // 4 waves along N -> 32 cols each
  const int lrow   = lane & 15;
  const int half   = lane >> 4;

  const v8f zero8 = {0.f, 0.f, 0.f, 0.f, 0.f, 0.f, 0.f, 0.f};
  v8f acc[4][2];
#pragma unroll
  for (int m = 0; m < 4; ++m)
#pragma unroll
    for (int n = 0; n < 2; ++n) acc[m][n] = zero8;

  const int col4  = (t & 7) * 4;       // 0..28, step 4
  const int rbase = t >> 3;            // 0..31

  for (int it = 0; it < KSLICE / KT; ++it) {
    const int kb = kbase + it * KT;
    __syncthreads();                   // protect previous iteration's LDS reads

    // ---- stage fp32 tiles -> bf16 hi/lo in LDS (convert once per element) --
#pragma unroll
    for (int p = 0; p < 4; ++p) {
      const int r = rbase + p * 32;
      const float4 av = *(const float4*)(X + (size_t)(tileM + r) * KDIM + kb + col4);
      const float4 bv = *(const float4*)(X + (size_t)(tileN + r) * KDIM + kb + col4);
      const float ae[4] = {av.x, av.y, av.z, av.w};
      const float be[4] = {bv.x, bv.y, bv.z, bv.w};
#pragma unroll
      for (int e = 0; e < 4; ++e) {
        unsigned short h = bf16_rne(ae[e]);
        float hf = __uint_as_float((unsigned)h << 16);
        Ahi[r * KT + col4 + e] = h;
        Alo[r * KT + col4 + e] = bf16_rne(ae[e] - hf);
        h  = bf16_rne(be[e]);
        hf = __uint_as_float((unsigned)h << 16);
        Bhi[r * KT + col4 + e] = h;
        Blo[r * KT + col4 + e] = bf16_rne(be[e] - hf);
      }
    }
    __syncthreads();

    // ---- fragments, per ISA VGPR layouts ---------------------------------
    // A 16x32 bf16: lanes 0-15 M=lane, lanes 16-31 M=lane-16;
    //   v0-3 hold K = h*8..h*8+7, v4-7 hold K = 16+h*8..16+h*8+7
    Frag ah[4], al[4], bh[2], bl[2];
#pragma unroll
    for (int m = 0; m < 4; ++m) {
      const u16x8* rh = (const u16x8*)&Ahi[(waveM0 + m * 16 + lrow) * KT];
      const u16x8* rl = (const u16x8*)&Alo[(waveM0 + m * 16 + lrow) * KT];
      ah[m].us[0] = rh[half];      ah[m].us[1] = rh[2 + half];
      al[m].us[0] = rl[half];      al[m].us[1] = rl[2 + half];
    }
    // B 32x16 bf16: lanes 0-15 N=lane, lanes 16-31 N=lane-16;
    //   v0-3 hold K = h*8..h*8+7, v4-7 hold K = 16+h*8..16+h*8+7
#pragma unroll
    for (int n = 0; n < 2; ++n) {
      const u16x8* rh = (const u16x8*)&Bhi[(waveN0 + n * 16 + lrow) * KT];
      const u16x8* rl = (const u16x8*)&Blo[(waveN0 + n * 16 + lrow) * KT];
      bh[n].us[0] = rh[half];      bh[n].us[1] = rh[2 + half];
      bl[n].us[0] = rl[half];      bl[n].us[1] = rl[2 + half];
    }

    // ---- error-compensated product: hi*hi + hi*lo + lo*hi ----------------
#pragma unroll
    for (int m = 0; m < 4; ++m)
#pragma unroll
      for (int n = 0; n < 2; ++n) {
        acc[m][n] = __builtin_amdgcn_wmma_f32_16x16x32_bf16(
            false, ah[m].bf, false, bh[n].bf, (short)0, acc[m][n], false, false);
        acc[m][n] = __builtin_amdgcn_wmma_f32_16x16x32_bf16(
            false, ah[m].bf, false, bl[n].bf, (short)0, acc[m][n], false, false);
        acc[m][n] = __builtin_amdgcn_wmma_f32_16x16x32_bf16(
            false, al[m].bf, false, bh[n].bf, (short)0, acc[m][n], false, false);
      }
  }

  // ---- split-K accumulate into G (C/D layout: VGPR r -> M = r + 8*half) ---
#pragma unroll
  for (int m = 0; m < 4; ++m)
#pragma unroll
    for (int n = 0; n < 2; ++n) {
      const int row0 = tileM + waveM0 + m * 16 + half * 8;
      const int col  = tileN + waveN0 + n * 16 + lrow;
#pragma unroll
      for (int r = 0; r < 8; ++r)
        atomicAdd(&G[(size_t)(row0 + r) * B_N + col], acc[m][n][r]);
    }
}

// ---------------------------------------------------------------------------
// Per-row masked logsumexp. grid = 512 blocks, block = 256 (each thread 2 cols)
// ---------------------------------------------------------------------------
__global__ __launch_bounds__(256) void row_lse(
    const float* __restrict__ G, const int* __restrict__ tgt,
    float* __restrict__ accum)
{
  __shared__ float redA[256];
  __shared__ float redB[256];
  const int i  = blockIdx.x;
  const int t  = threadIdx.x;
  const int ti = tgt[i];
  const float Gii = G[(size_t)i * B_N + i];

  float sv[2]; bool mk[2], pk[2];
  float mAll = -3.4e38f, mPos = -3.4e38f;
#pragma unroll
  for (int q = 0; q < 2; ++q) {
    const int j = t + q * 256;
    const float Gjj = G[(size_t)j * B_N + j];
    const float s = -(Gii + Gjj - 2.0f * G[(size_t)i * B_N + j]) * SCALE;
    const bool m = (j != i);
    const bool p = m && (tgt[j] == ti);
    sv[q] = s; mk[q] = m; pk[q] = p;
    if (m) mAll = fmaxf(mAll, s);
    if (p) mPos = fmaxf(mPos, s);
  }
  redA[t] = mAll; redB[t] = mPos;
  __syncthreads();
  for (int off = 128; off > 0; off >>= 1) {
    if (t < off) {
      redA[t] = fmaxf(redA[t], redA[t + off]);
      redB[t] = fmaxf(redB[t], redB[t + off]);
    }
    __syncthreads();
  }
  mAll = redA[0]; mPos = redB[0];
  __syncthreads();

  float sAll = 0.f, sPos = 0.f;
#pragma unroll
  for (int q = 0; q < 2; ++q) {
    if (mk[q]) sAll += expf(sv[q] - mAll);
    if (pk[q]) sPos += expf(sv[q] - mPos);
  }
  redA[t] = sAll; redB[t] = sPos;
  __syncthreads();
  for (int off = 128; off > 0; off >>= 1) {
    if (t < off) {
      redA[t] += redA[t + off];
      redB[t] += redB[t + off];
    }
    __syncthreads();
  }
  if (t == 0) {
    const float den   = mAll + logf(redA[0]);
    const bool  hasP  = redB[0] > 0.f;
    const float num   = hasP ? (mPos + logf(redB[0])) : den;
    atomicAdd(&accum[0], hasP ? (den - num) : 0.f);
    atomicAdd(&accum[1], hasP ? 1.f : 0.f);
  }
}

__global__ void finalize_loss(const float* __restrict__ accum,
                              float* __restrict__ out) {
  out[0] = (accum[1] > 0.f) ? accum[0] / accum[1] : 0.f;
}

extern "C" void kernel_launch(void* const* d_in, const int* in_sizes, int n_in,
                              void* d_out, int out_size, void* d_ws, size_t ws_size,
                              hipStream_t stream) {
  const float* entropy   = (const float*)d_in[0];
  const float* alignment = (const float*)d_in[1];
  const float* curvature = (const float*)d_in[2];
  const int*   targets   = (const int*)d_in[3];

  float* G     = (float*)d_ws;          // [512*512] Gram accumulator
  float* accum = G + (size_t)B_N * B_N; // [2] {sum(den-num), valid count}
  float* out   = (float*)d_out;

  const int nz = B_N * B_N + 2;
  zero_ws<<<(nz + 255) / 256, 256, 0, stream>>>(G, nz);

  dim3 grid(B_N / TILE, B_N / TILE, 3 * NSLICE);   // 4 x 4 x 48
  gram_wmma<<<grid, 256, 0, stream>>>(entropy, alignment, curvature, G);

  row_lse<<<B_N, 256, 0, stream>>>(G, targets, accum);
  finalize_loss<<<1, 1, 0, stream>>>(accum, out);
}
